// RelationalGraphConvolution_89318139887861
// MI455X (gfx1250) — matrix operand order repaired and checked by reading, not compiled, not run on hardware
//
#include <hip/hip_runtime.h>
#include <hip/hip_bf16.h>

// ---------------------------------------------------------------------------
// RelationalGraphConvolution for MI455X (gfx1250, wave32, WMMA)
//
// out[n,u] = relu( sum_r (A_r @ (X @ W_r))[n,u] + (X @ S)[n,u] + bias[u] )
// N=8192, F=U=64, R=4, B=20.  Dominant cost: sum_r A_r @ xw_r
//   = 34.4 GFLOP over 1.07 GB of adjacency  (AI ~ 32 FLOP/B -> HBM bound,
//   floor ~46us at 23.3 TB/s).  xw in bf16 is 4 MB -> L2 resident (192 MB).
// Strategy: stream A once, cvt fp32->bf16 in regs, v_wmma_f32_16x16x32_bf16,
// software-pipelined; each wave owns a 32-row tile so every B fragment from
// L2 feeds two WMMAs (halves L2 hit traffic vs a 16-row tile).
// ---------------------------------------------------------------------------

#define N_NODES 8192
#define F_DIM   64
#define U_DIM   64
#define R_REL   4
#define B_BASES 20
#define KB_CNT  (N_NODES / 32)     // 256 K-blocks of 32 along the m dimension

typedef __attribute__((ext_vector_type(16))) __bf16 v16bf;
typedef __attribute__((ext_vector_type(8)))  float  v8f;

// Workspace layout (bytes):
//   [0,      65536)                       W:     R*F*U fp32
//   [65536,  65536+4194304)               panel: xw in bf16, B-frag swizzled
//   [65536+4194304, +2097152)             selfx: N*U fp32 (X@S + bias)
#define WS_W_OFF      0
#define WS_PANEL_OFF  65536
#define WS_SELFX_OFF  (65536 + 4194304)

// --------------------------- Kernel 0: W = bases . coeff -------------------
__global__ __launch_bounds__(256) void wcomb_kernel(
    const float* __restrict__ bases,    // (F,U,B)
    const float* __restrict__ coeff,    // (R,B)
    float* __restrict__ W)              // (R,F,U)
{
    int idx = blockIdx.x * 256 + threadIdx.x;
    if (idx >= R_REL * F_DIM * U_DIM) return;
    int r  = idx / (F_DIM * U_DIM);
    int fu = idx % (F_DIM * U_DIM);
    float acc = 0.f;
    #pragma unroll
    for (int k = 0; k < B_BASES; ++k)
        acc += bases[fu * B_BASES + k] * coeff[r * B_BASES + k];
    W[idx] = acc;
}

// ------------- Kernel 1: xw = X @ W_r (bf16, B-frag layout) + self ---------
// Panel layout: for (r, kb, ct):  chunk of 1024 B at ((r*KB+kb)*4+ct)*1024.
// Within a chunk, lane L stores 16 bf16 (32 B) = B[k][col] with
//   col = L%16, k = (L/16)*8 + e (e<8), 16 + (L/16)*8 + (e-8) (e>=8)  -> matches the
//   32x16 16-bit B operand layout consumed by v_wmma_f32_16x16x32_bf16.
__global__ __launch_bounds__(256) void xw_pack_kernel(
    const float* __restrict__ x,        // (N,F)
    const float* __restrict__ W,        // (R,F,U)
    const float* __restrict__ self_k,   // (F,U)
    const float* __restrict__ bias,     // (U)
    __bf16* __restrict__ panel,
    float* __restrict__ selfx)          // (N,U)
{
    __shared__ float xls[32 * 65];      // padded: kills 32-way bank conflicts
    __shared__ float wls[64 * 64];

    int rr = blockIdx.x / KB_CNT;       // 0..R_REL  (R_REL == self term)
    int kb = blockIdx.x % KB_CNT;
    int t  = threadIdx.x;
    int m0 = kb * 32;

    // stage 32x64 rows of X (contiguous) and the 64x64 weight matrix
    #pragma unroll
    for (int i = 0; i < 8; ++i) {
        int idx = t + i * 256;
        xls[(idx >> 6) * 65 + (idx & 63)] = x[(size_t)m0 * F_DIM + idx];
    }
    const float* wsrc = (rr < R_REL) ? (W + rr * F_DIM * U_DIM) : self_k;
    #pragma unroll
    for (int i = 0; i < 16; ++i)
        wls[t + i * 256] = wsrc[t + i * 256];
    __syncthreads();

    int u0 = (t >> 5) * 8;              // 8 consecutive u per thread
    int m  = t & 31;                    // local k (row of xw block)
    float acc[8] = {};
    for (int f = 0; f < F_DIM; ++f) {
        float xv = xls[m * 65 + f];
        #pragma unroll
        for (int j = 0; j < 8; ++j)
            acc[j] += xv * wls[f * U_DIM + u0 + j];
    }

    if (rr < R_REL) {
        #pragma unroll
        for (int j = 0; j < 8; ++j) {
            int u    = u0 + j;
            int ct   = u >> 4;
            int col  = u & 15;
            int lane = col + ((m & 8) ? 16 : 0);
            int e    = (m & 7) + ((m & 16) >> 1);
            size_t off = ((size_t)((rr * KB_CNT + kb) * 4 + ct) << 9)  // *512 elems
                       + (size_t)lane * 16 + e;
            panel[off] = (__bf16)acc[j];
        }
    } else {
        #pragma unroll
        for (int j = 0; j < 8; ++j) {
            int u = u0 + j;
            selfx[(size_t)(m0 + m) * U_DIM + u] = acc[j] + bias[u];
        }
    }
}

// ---------------------- agg kernel helper fragments ------------------------
struct AFrag { float4 a0, a1, a2, a3; };
struct BFrag { v16bf b0, b1, b2, b3; };

__device__ __forceinline__ AFrag load_a(const float* __restrict__ Arow,
                                        int m0, int half) {
    AFrag f;
    const float4* p1 = (const float4*)(Arow + m0 + half * 8);
    f.a0 = p1[0];
    f.a1 = p1[1];
    const float4* p2 = (const float4*)(Arow + m0 + 16 + half * 8);
    f.a2 = p2[0];
    f.a3 = p2[1];
    return f;
}

__device__ __forceinline__ BFrag load_b(const __bf16* __restrict__ pb) {
    BFrag f;
    f.b0 = *(const v16bf*)(pb);
    f.b1 = *(const v16bf*)(pb + 512);
    f.b2 = *(const v16bf*)(pb + 1024);
    f.b3 = *(const v16bf*)(pb + 1536);
    return f;
}

__device__ __forceinline__ v16bf cvt_a(const AFrag& f) {
    v16bf a;
    a[0]=(__bf16)f.a0.x;  a[1]=(__bf16)f.a0.y;  a[2]=(__bf16)f.a0.z;  a[3]=(__bf16)f.a0.w;
    a[4]=(__bf16)f.a1.x;  a[5]=(__bf16)f.a1.y;  a[6]=(__bf16)f.a1.z;  a[7]=(__bf16)f.a1.w;
    a[8]=(__bf16)f.a2.x;  a[9]=(__bf16)f.a2.y;  a[10]=(__bf16)f.a2.z; a[11]=(__bf16)f.a2.w;
    a[12]=(__bf16)f.a3.x; a[13]=(__bf16)f.a3.y; a[14]=(__bf16)f.a3.z; a[15]=(__bf16)f.a3.w;
    return a;
}

__device__ __forceinline__ void mma4(v8f c[4], v16bf a, const BFrag& bf) {
    c[0] = __builtin_amdgcn_wmma_f32_16x16x32_bf16(false, a, false, bf.b0,
                                                   (short)0, c[0], false, false);
    c[1] = __builtin_amdgcn_wmma_f32_16x16x32_bf16(false, a, false, bf.b1,
                                                   (short)0, c[1], false, false);
    c[2] = __builtin_amdgcn_wmma_f32_16x16x32_bf16(false, a, false, bf.b2,
                                                   (short)0, c[2], false, false);
    c[3] = __builtin_amdgcn_wmma_f32_16x16x32_bf16(false, a, false, bf.b3,
                                                   (short)0, c[3], false, false);
}

// ------------- Kernel 2: agg = sum_r A_r @ xw_r ; + self ; relu ------------
// 1 block = 32 output rows; 8 waves. Wave w = (relation r = w&3,
// K-half kh = w>>2): sweeps 128 K-blocks of 32 over rows n0..n0+31.
// Each B fragment (L2) feeds TWO WMMAs (row-tiles n0.. and n0+16..), so L2
// hit traffic is ~1 GB instead of ~2 GB on top of the 1.07 GB HBM stream.
// Software-pipelined: loads for step kb+1 issue before the WMMAs of step kb;
// prefetch runs 8 K-blocks ahead (offset wrapped into the row: branch-free,
// in-bounds). 8 partials reduced through LDS, + self term, ReLU.
__global__ __launch_bounds__(256) void agg_kernel(
    const float*  __restrict__ A,       // (R,N,N)
    const __bf16* __restrict__ panel,
    const float*  __restrict__ selfx,   // (N,U)
    float* __restrict__ out)            // (N,U)
{
    __shared__ float red[2 * R_REL][32][U_DIM];   // 64 KB

    int n0   = blockIdx.x * 32;
    int w    = threadIdx.x >> 5;
    int r    = w & 3;                   // relation
    int kh   = w >> 2;                  // K-half (0:first 128 kb, 1:second)
    int lane = threadIdx.x & 31;
    int lo   = lane & 15;               // A row (M), C column
    int half = lane >> 4;               // K-octet selector / C row-half

    const float*  Arow0 = A + ((size_t)r * N_NODES + n0 + lo) * N_NODES;
    const float*  Arow1 = Arow0 + (size_t)16 * N_NODES;   // rows n0+16..n0+31
    const __bf16* pb0   = panel + (((size_t)r * KB_CNT * 4) << 9)
                                + (size_t)lane * 16;

    int kb_lo = kh * (KB_CNT / 2);
    int kb_hi = kb_lo + (KB_CNT / 2);

    v8f c0[4], c1[4];
    #pragma unroll
    for (int ct = 0; ct < 4; ++ct) {
        c0[ct] = (v8f){0.f,0.f,0.f,0.f,0.f,0.f,0.f,0.f};
        c1[ct] = (v8f){0.f,0.f,0.f,0.f,0.f,0.f,0.f,0.f};
    }

    // ---- prologue: fragments for the first K-step of this wave's range
    AFrag af0 = load_a(Arow0, kb_lo * 32, half);
    AFrag af1 = load_a(Arow1, kb_lo * 32, half);
    BFrag bf  = load_b(pb0 + ((size_t)kb_lo << 11));

    // ---- steady state: next loads first, then math on current fragments
    for (int kb = kb_lo; kb < kb_hi - 1; ++kb) {
        AFrag afn0 = load_a(Arow0, (kb + 1) * 32, half);
        AFrag afn1 = load_a(Arow1, (kb + 1) * 32, half);
        BFrag bfn  = load_b(pb0 + ((size_t)(kb + 1) << 11));  // +2048 elems/kb
        int pfc = (kb * 32 + 256) & (N_NODES - 1);
        __builtin_prefetch(Arow0 + pfc, 0, 1);
        __builtin_prefetch(Arow1 + pfc, 0, 1);
        v16bf a0 = cvt_a(af0);
        v16bf a1 = cvt_a(af1);
        mma4(c0, a0, bf);     // same B fragment feeds both row-tiles
        mma4(c1, a1, bf);
        af0 = afn0;
        af1 = afn1;
        bf  = bfn;
    }
    // ---- epilogue: last K-step of this wave's range
    {
        v16bf a0 = cvt_a(af0);
        v16bf a1 = cvt_a(af1);
        mma4(c0, a0, bf);
        mma4(c1, a1, bf);
    }

    // C layout: VGPR j of lane L holds C[M = j + 8*(L>>4)][u = ct*16 + (L&15)]
    #pragma unroll
    for (int j = 0; j < 8; ++j) {
        int mrow = j + 8 * half;
        red[w][mrow     ][0 * 16 + lo] = c0[0][j];
        red[w][mrow     ][1 * 16 + lo] = c0[1][j];
        red[w][mrow     ][2 * 16 + lo] = c0[2][j];
        red[w][mrow     ][3 * 16 + lo] = c0[3][j];
        red[w][mrow + 16][0 * 16 + lo] = c1[0][j];
        red[w][mrow + 16][1 * 16 + lo] = c1[1][j];
        red[w][mrow + 16][2 * 16 + lo] = c1[2][j];
        red[w][mrow + 16][3 * 16 + lo] = c1[3][j];
    }
    __syncthreads();

    #pragma unroll
    for (int i = 0; i < 8; ++i) {
        int idx = threadIdx.x + i * 256;
        int nl = idx >> 6, u = idx & 63;
        float v = red[0][nl][u] + red[1][nl][u] + red[2][nl][u] + red[3][nl][u]
                + red[4][nl][u] + red[5][nl][u] + red[6][nl][u] + red[7][nl][u]
                + selfx[(size_t)(n0 + nl) * U_DIM + u];
        out[(size_t)(n0 + nl) * U_DIM + u] = v > 0.f ? v : 0.f;
    }
}

// ---------------------------------------------------------------------------
extern "C" void kernel_launch(void* const* d_in, const int* in_sizes, int n_in,
                              void* d_out, int out_size, void* d_ws, size_t ws_size,
                              hipStream_t stream) {
    const float* features     = (const float*)d_in[0];   // (1,N,F)
    // d_in[1] = out_indices (int64) -- unused: final_layer=False returns all nodes
    const float* adjacency    = (const float*)d_in[2];   // (R,N,N)
    const float* bases        = (const float*)d_in[3];   // (F,U,B)
    const float* coefficients = (const float*)d_in[4];   // (R,B)
    const float* self_kernel  = (const float*)d_in[5];   // (F,U)
    const float* bias         = (const float*)d_in[6];   // (U,)
    float* out = (float*)d_out;

    char*   ws    = (char*)d_ws;
    float*  W     = (float*)(ws + WS_W_OFF);
    __bf16* panel = (__bf16*)(ws + WS_PANEL_OFF);
    float*  selfx = (float*)(ws + WS_SELFX_OFF);

    wcomb_kernel<<<(R_REL * F_DIM * U_DIM + 255) / 256, 256, 0, stream>>>(
        bases, coefficients, W);
    xw_pack_kernel<<<(R_REL + 1) * KB_CNT, 256, 0, stream>>>(
        features, W, self_kernel, bias, panel, selfx);
    agg_kernel<<<N_NODES / 32, 256, 0, stream>>>(
        adjacency, panel, selfx, out);
}
